// RNN_70145405878815
// MI455X (gfx1250) — matrix operand order repaired and checked by reading, not compiled, not run on hardware
//
#include <hip/hip_runtime.h>

#define T_STEPS 8192
#define IN_DIM  1024
#define HD      2048   // HD_RNN
#define TOF     1024   // TO_FFNN
#define HDF     2048   // HD_FFNN
#define NCLS    5

typedef __bf16 bf16;
typedef bf16  v16bf __attribute__((ext_vector_type(16)));
typedef bf16  v8bf  __attribute__((ext_vector_type(8)));
typedef float v8f   __attribute__((ext_vector_type(8)));

#define SCAN_BLOCKS 64
#define ROW_PAD     8                     // bf16 elements of pad per LDS row
#define LDS_ROW     (HD + ROW_PAD)        // 2056 bf16 = 4112 B (16B aligned)
#define SCAN_LDS_BYTES (32 * LDS_ROW * 2) // 131584 B per block

// ---------------- fp32 -> bf16 conversion (grid-stride) ----------------
__global__ void cvt_f32_bf16(const float* __restrict__ src,
                             bf16* __restrict__ dst, int n) {
  int i = blockIdx.x * blockDim.x + threadIdx.x;
  int stride = gridDim.x * blockDim.x;
  for (; i < n; i += stride) dst[i] = (bf16)src[i];
}

// ---------------- W [2048,1024] fp32 -> Wt [1024,2048] bf16 --------------
__global__ void cvt_transpose_w(const float* __restrict__ W,
                                bf16* __restrict__ Wt) {
  int idx = blockIdx.x * blockDim.x + threadIdx.x;
  int stride = gridDim.x * blockDim.x;
  for (; idx < HD * IN_DIM; idx += stride) {
    const int k = idx / HD;        // write-coalesced over Wt
    const int n = idx % HD;
    Wt[idx] = (bf16)W[(size_t)n * IN_DIM + k];
  }
}

// ---------------- zero h0 (fp32 + bf16) and sync counter ----------------
__global__ void init_state(float* __restrict__ hf, bf16* __restrict__ hb,
                           unsigned* __restrict__ cnt) {
  int i = blockIdx.x * blockDim.x + threadIdx.x;
  if (i < HD) { hf[i] = 0.f; hb[i] = (bf16)0.f; }
  if (i == 0) *cnt = 0u;
}

// ---------------- Phase 1: wx = X @ W^T + Wb  (bf16 WMMA, f32 acc) -------
// X: [8192,1024] bf16 row-major, Wt: [1024,2048] bf16 (W transposed).
// Block: 256 thr = 8 waves. Wave covers 4 M-tiles (64 rows) x 1 N-tile.
// Grid: (2048/16, 8192/512) = (128, 16)
__global__ void __launch_bounds__(256)
gemm_wx(const bf16* __restrict__ X, const bf16* __restrict__ Wt,
        const float* __restrict__ Wb, float* __restrict__ wx) {
  const int lane  = threadIdx.x & 31;
  const int wave  = threadIdx.x >> 5;
  const int n0    = blockIdx.x * 16;
  const int mt0   = blockIdx.y * 32 + wave * 4;   // first of 4 M-tiles
  const int arow  = lane & 15;                    // A: M within tile
  const int akoff = (lane >> 4) << 3;             // A: K sub-offset (0 or 8)

  v8f acc[4] = {};
  for (int k0 = 0; k0 < IN_DIM; k0 += 32) {
    // B fragment: lane%16 = N, elements = K (akoff+e, akoff+16+e) from Wt columns
    const bf16* bp = Wt + (size_t)(k0 + akoff) * HD + n0 + arow;
    v16bf b;
#pragma unroll
    for (int e = 0; e < 8; ++e) { b[e] = bp[(size_t)e * HD]; b[8 + e] = bp[(size_t)(16 + e) * HD]; }

#pragma unroll
    for (int i = 0; i < 4; ++i) {
      const bf16* ap = X + (size_t)((mt0 + i) * 16 + arow) * IN_DIM + k0 + akoff;
      v8bf lo = *(const v8bf*)ap;          // K = akoff..akoff+7
      v8bf hi = *(const v8bf*)(ap + 16);   // K = akoff+16..akoff+23
      v16bf a;
#pragma unroll
      for (int e = 0; e < 8; ++e) { a[e] = lo[e]; a[8 + e] = hi[e]; }
      acc[i] = __builtin_amdgcn_wmma_f32_16x16x32_bf16(
          false, a, false, b, (short)0, acc[i], false, false);
    }
  }
  // D layout: lane 0-15 -> (M=r, N=lane); lane 16-31 -> (M=8+r, N=lane-16)
  const int   n    = n0 + (lane & 15);
  const float bias = Wb[n];
  const int   msub = (lane >> 4) << 3;
#pragma unroll
  for (int i = 0; i < 4; ++i) {
    const int mrow = (mt0 + i) * 16 + msub;
#pragma unroll
    for (int r = 0; r < 8; ++r)
      wx[(size_t)(mrow + r) * HD + n] = acc[i][r] + bias;
  }
}

// ---------------- Phase 2: persistent recurrent scan ----------------
// h_t = relu(U @ h_{t-1} + Ub + wx[t]).
// 64 blocks x 2 waves; each block stages its 32-row U slice into LDS once
// (async DMA), then every step runs the matvec out of LDS via WMMA with
// B = h broadcast across all N columns. Four independent accumulators break
// the WMMA dependency chain (64 -> 16 deep). Device-wide step barrier via
// monotonic atomic counter.
__global__ void __launch_bounds__(64)
rnn_scan(const bf16* __restrict__ U, const float* __restrict__ Ub,
         const float* __restrict__ wx, float* __restrict__ hf,
         bf16* __restrict__ hb, unsigned* __restrict__ cnt) {
  extern __shared__ bf16 Us[];            // [32][LDS_ROW]
  const int tid   = threadIdx.x;
  const int lane  = tid & 31;
  const int wave  = tid >> 5;
  const int R0    = blockIdx.x * 32;      // first global row owned by block
  const int arow  = lane & 15;
  const int akoff = (lane >> 4) << 3;
  const unsigned nb = gridDim.x;

  // ---- stage U slice (32 rows x 2048 bf16) into padded LDS, async DMA ----
  {
    const char* gbase = (const char*)(U + (size_t)R0 * HD);
    for (int c = tid; c < 32 * (HD * 2 / 16); c += 64) {
      const int row = c >> 8;            // 256 x 16B chunks per row
      const int col = c & 255;
      unsigned lds_addr =
          (unsigned)(uintptr_t)(Us + (size_t)row * LDS_ROW) + (unsigned)(col * 16);
      unsigned long long ga =
          (unsigned long long)(uintptr_t)(gbase + (size_t)row * (HD * 2) + col * 16);
      asm volatile("global_load_async_to_lds_b128 %0, %1, off"
                   :: "v"(lds_addr), "v"(ga) : "memory");
    }
    asm volatile("s_wait_asynccnt 0x0" ::: "memory");
  }
  __syncthreads();

  const bf16* lrow = Us + (size_t)(wave * 16 + arow) * LDS_ROW + akoff;
  const int mb = R0 + wave * 16 + ((lane >> 4) << 3);
  float biasv[8];
#pragma unroll
  for (int r = 0; r < 8; ++r) biasv[r] = Ub[mb + r];

  for (int t = 0; t < T_STEPS; ++t) {
    const unsigned short* hcu =
        (const unsigned short*)(hb + (size_t)(t & 1) * HD);
    v8f acc0 = {}, acc1 = {}, acc2 = {}, acc3 = {};

#define SCAN_CHUNK(KO, ACC)                                                  \
    {                                                                        \
      const unsigned* hp = (const unsigned*)(hcu + (KO) + akoff);            \
      union { unsigned u[8]; v16bf v; } bb;                                  \
      _Pragma("unroll")                                                      \
      for (int j = 0; j < 4; ++j) { bb.u[j] = hp[j]; bb.u[4 + j] = hp[8 + j]; } \
      v8bf lo = *(const v8bf*)(lrow + (KO));                                 \
      v8bf hi = *(const v8bf*)(lrow + (KO) + 16);                            \
      v16bf a;                                                               \
      _Pragma("unroll")                                                      \
      for (int e = 0; e < 8; ++e) { a[e] = lo[e]; a[8 + e] = hi[e]; }        \
      ACC = __builtin_amdgcn_wmma_f32_16x16x32_bf16(                         \
          false, a, false, bb.v, (short)0, ACC, false, false);               \
    }

#pragma unroll 2
    for (int k0 = 0; k0 < HD; k0 += 128) {   // 4 independent chunks / iter
      SCAN_CHUNK(k0,      acc0)
      SCAN_CHUNK(k0 + 32, acc1)
      SCAN_CHUNK(k0 + 64, acc2)
      SCAN_CHUNK(k0 + 96, acc3)
    }
#undef SCAN_CHUNK
    const v8f acc = (acc0 + acc1) + (acc2 + acc3);

    // every N column of D is identical; lanes with N==0 commit 8 rows each
    if ((lane & 15) == 0) {
      float* hfn = hf + (size_t)((t + 1) & 1) * HD;
      bf16*  hbn = hb + (size_t)((t + 1) & 1) * HD;
      const float* wxt = wx + (size_t)t * HD;
#pragma unroll
      for (int r = 0; r < 8; ++r) {
        float v = acc[r] + biasv[r] + wxt[mb + r];
        v = v > 0.f ? v : 0.f;
        hfn[mb + r] = v;
        hbn[mb + r] = (bf16)v;
      }
    }
    // ---- device-wide step barrier (monotonic counter) ----
    __threadfence();
    __syncthreads();
    if (tid == 0) {
      __hip_atomic_fetch_add(cnt, 1u, __ATOMIC_RELEASE, __HIP_MEMORY_SCOPE_AGENT);
      const unsigned target = (unsigned)(t + 1) * nb;
      while (__hip_atomic_load(cnt, __ATOMIC_ACQUIRE, __HIP_MEMORY_SCOPE_AGENT) < target)
        __builtin_amdgcn_s_sleep(1);
    }
    __syncthreads();
  }
}

// ---------------- Phase 3: head (3 small matvecs + log_softmax) ----------
__global__ void __launch_bounds__(256)
head_kernel(const float* __restrict__ hT,
            const float* __restrict__ Vw,  const float* __restrict__ Vb,
            const float* __restrict__ W1w, const float* __restrict__ W1b,
            const float* __restrict__ W2w, const float* __restrict__ W2b,
            float* __restrict__ out) {
  __shared__ float y[TOF];
  __shared__ float z1[HDF];
  __shared__ float z2[NCLS];
  const int tid = threadIdx.x;

  for (int j = tid; j < TOF; j += 256) {
    float acc = Vb[j];
    const float* vr = Vw + (size_t)j * HD;
    for (int k = 0; k < HD; ++k) acc += vr[k] * hT[k];
    y[j] = acc > 0.f ? acc : 0.f;
  }
  __syncthreads();
  for (int i = tid; i < HDF; i += 256) {
    float acc = W1b[i];
    const float* wr = W1w + (size_t)i * TOF;
    for (int k = 0; k < TOF; ++k) acc += wr[k] * y[k];
    z1[i] = acc > 0.f ? acc : 0.f;
  }
  __syncthreads();
  if (tid < NCLS) {
    float acc = W2b[tid];
    const float* wr = W2w + (size_t)tid * HDF;
    for (int k = 0; k < HDF; ++k) acc += wr[k] * z1[k];
    z2[tid] = acc;
  }
  __syncthreads();
  if (tid == 0) {
    float m = z2[0];
    for (int c = 1; c < NCLS; ++c) m = z2[c] > m ? z2[c] : m;
    float s = 0.f;
    for (int c = 0; c < NCLS; ++c) s += __expf(z2[c] - m);
    const float l = __logf(s);
    for (int c = 0; c < NCLS; ++c) out[c] = z2[c] - m - l;
  }
}

extern "C" void kernel_launch(void* const* d_in, const int* in_sizes, int n_in,
                              void* d_out, int out_size, void* d_ws, size_t ws_size,
                              hipStream_t stream) {
  (void)in_sizes; (void)n_in; (void)out_size; (void)ws_size;
  const float* inputs = (const float*)d_in[0];
  const float* U_w  = (const float*)d_in[1];
  const float* U_b  = (const float*)d_in[2];
  const float* W_w  = (const float*)d_in[3];
  const float* W_b  = (const float*)d_in[4];
  const float* V_w  = (const float*)d_in[5];
  const float* V_b  = (const float*)d_in[6];
  const float* W1_w = (const float*)d_in[7];
  const float* W1_b = (const float*)d_in[8];
  const float* W2_w = (const float*)d_in[9];
  const float* W2_b = (const float*)d_in[10];
  float* out = (float*)d_out;

  char* ws = (char*)d_ws;
  size_t o = 0;
  bf16*  Xbf = (bf16*)(ws + o);  o += (size_t)T_STEPS * IN_DIM * 2;  // 16 MB
  bf16*  Wt  = (bf16*)(ws + o);  o += (size_t)HD * IN_DIM * 2;       //  4 MB
  bf16*  Ubf = (bf16*)(ws + o);  o += (size_t)HD * HD * 2;           //  8 MB
  float* wx  = (float*)(ws + o); o += (size_t)T_STEPS * HD * 4;      // 64 MB
  float* hf  = (float*)(ws + o); o += (size_t)2 * HD * 4;            // double buffer
  bf16*  hb  = (bf16*)(ws + o);  o += (size_t)2 * HD * 2;
  unsigned* cnt = (unsigned*)(ws + o);

  (void)hipFuncSetAttribute(reinterpret_cast<const void*>(rnn_scan),
                            hipFuncAttributeMaxDynamicSharedMemorySize,
                            SCAN_LDS_BYTES);

  cvt_f32_bf16<<<2048, 256, 0, stream>>>(inputs, Xbf, T_STEPS * IN_DIM);
  cvt_transpose_w<<<1024, 256, 0, stream>>>(W_w, Wt);
  cvt_f32_bf16<<<2048, 256, 0, stream>>>(U_w, Ubf, HD * HD);
  init_state<<<(HD + 255) / 256, 256, 0, stream>>>(hf, hb, cnt);

  gemm_wx<<<dim3(HD / 16, T_STEPS / 512), 256, 0, stream>>>(Xbf, Wt, W_b, wx);
  rnn_scan<<<SCAN_BLOCKS, 64, SCAN_LDS_BYTES, stream>>>(Ubf, U_b, wx, hf, hb, cnt);
  // final h lands in buffer (T_STEPS & 1) == 0
  head_kernel<<<1, 256, 0, stream>>>(hf, V_w, V_b, W1_w, W1_b, W2_w, W2_b, out);
}
